// MambaBlock_76304388980859
// MI455X (gfx1250) — matrix operand order repaired and checked by reading, not compiled, not run on hardware
//
#include <hip/hip_runtime.h>
#include <hip/hip_bf16.h>

// ---------------- types for CDNA5 WMMA ----------------
typedef __attribute__((ext_vector_type(16))) __bf16 v16bf;
typedef __attribute__((ext_vector_type(8)))  float  v8f;
typedef __attribute__((ext_vector_type(4)))  int    v4i;

struct AFrag { union { v16bf v; uint4 q[2]; }; };
struct BFrag { union { v16bf v; uint4 q[2]; }; };

static __device__ inline v8f v8f_zero() {
    v8f z = {0.f,0.f,0.f,0.f,0.f,0.f,0.f,0.f};
    return z;
}
// A tile 16x32 bf16: base = A + row*lda + hf*8 (+k0); chunks at +0 and +16 elems
static __device__ inline void load_a(AFrag& f, const __bf16* p) {
    f.q[0] = *reinterpret_cast<const uint4*>(p);
    f.q[1] = *reinterpret_cast<const uint4*>(p + 16);
}
// B tile 32x16 bf16 (from row-major [N,K] weight): base = W + row*ldb + hf*8 (+k0); chunks at +0 and +16 elems
static __device__ inline void load_b(BFrag& f, const __bf16* p) {
    f.q[0] = *reinterpret_cast<const uint4*>(p);
    f.q[1] = *reinterpret_cast<const uint4*>(p + 16);
}
static __device__ inline v8f wmma_bf16(const AFrag& a, const BFrag& b, v8f c) {
    return __builtin_amdgcn_wmma_f32_16x16x32_bf16(
        false, a.v, false, b.v, (short)0, c, false, false);
}

static __device__ inline float silu_f(float x) { return x / (1.f + expf(-x)); }

// ---------------- async global -> LDS (CDNA5), with sync fallback ----------------
#if defined(__has_builtin)
#if __has_builtin(__builtin_amdgcn_global_load_async_to_lds_b128) && \
    __has_builtin(__builtin_amdgcn_s_wait_asynccnt)
#define USE_ASYNC_LDS 1
#endif
#endif

static __device__ inline void async_cp16(void* lds, const void* g) {
#ifdef USE_ASYNC_LDS
    __builtin_amdgcn_global_load_async_to_lds_b128(
        (__attribute__((address_space(1))) v4i*)g,
        (__attribute__((address_space(3))) v4i*)lds,
        0, 0);
#else
    *reinterpret_cast<uint4*>(lds) = *reinterpret_cast<const uint4*>(g);
#endif
}
static __device__ inline void async_wait_le4() {
#ifdef USE_ASYNC_LDS
    __builtin_amdgcn_s_wait_asynccnt(4);
#endif
}
static __device__ inline void async_wait_le0() {
#ifdef USE_ASYNC_LDS
    __builtin_amdgcn_s_wait_asynccnt(0);
#endif
}

// ---------------- problem sizes ----------------
#define BB   4
#define LL   1024
#define DMM  1024
#define DII  2048
#define NST  16
#define KCC  4
#define NTOK 4096   // BB*LL
#define LPITCH 40   // LDS row pitch in bf16 elems (32 data + 8 pad)

// ---------------- elementwise casts ----------------
__global__ __launch_bounds__(256) void k_cast_bf16(const float* __restrict__ src,
                                                   __bf16* __restrict__ dst, int n) {
    int i = blockIdx.x * 256 + threadIdx.x;
    if (i < n) dst[i] = (__bf16)src[i];
}

// conv_w [DI_out, DI_in, KC] -> 4 planes [kc][DI_out][DI_in] bf16
__global__ __launch_bounds__(256) void k_convw_rearrange(const float* __restrict__ src,
                                                         __bf16* __restrict__ dst) {
    int idx = blockIdx.x * 256 + threadIdx.x;           // o*DII + i
    if (idx >= DII * DII) return;
    float4 w = reinterpret_cast<const float4*>(src)[idx];
    const size_t plane = (size_t)DII * DII;
    dst[0 * plane + idx] = (__bf16)w.x;
    dst[1 * plane + idx] = (__bf16)w.y;
    dst[2 * plane + idx] = (__bf16)w.z;
    dst[3 * plane + idx] = (__bf16)w.w;
}

// ---------------- RMSNorm + bf16 cast ----------------
__global__ __launch_bounds__(256) void k_rmsnorm(const float* __restrict__ x,
                                                 const float* __restrict__ nw,
                                                 __bf16* __restrict__ xn) {
    __shared__ float red[256];
    const int row = blockIdx.x, tid = threadIdx.x;
    const size_t base = (size_t)row * DMM;
    float s = 0.f;
    for (int c = tid; c < DMM; c += 256) { float v = x[base + c]; s += v * v; }
    red[tid] = s; __syncthreads();
    for (int off = 128; off > 0; off >>= 1) {
        if (tid < off) red[tid] += red[tid + off];
        __syncthreads();
    }
    const float rms = rsqrtf(red[0] / (float)DMM + 1e-6f);
    for (int c = tid; c < DMM; c += 256)
        xn[base + c] = (__bf16)(x[base + c] * rms * nw[c]);
}

// ---------------- GEMM 1: xz = xn @ in_proj_w^T + b ; split ----------------
__global__ __launch_bounds__(256) void k_gemm_inproj(const __bf16* __restrict__ A,
                                                     const __bf16* __restrict__ W,
                                                     const float* __restrict__ bias,
                                                     __bf16* __restrict__ xssm,
                                                     float* __restrict__ gate) {
    const int lda = DMM, ldb = DMM, K = DMM;
    const int tid = threadIdx.x, wave = tid >> 5, lane = tid & 31;
    const int r = lane & 15, hf = lane >> 4;
    const int m0 = blockIdx.y * 128 + (wave >> 1) * 32;
    const int n0 = blockIdx.x * 128 + (wave & 1) * 64;

    v8f acc[2][4];
    for (int i = 0; i < 2; ++i) for (int j = 0; j < 4; ++j) acc[i][j] = v8f_zero();

    const __bf16* aP[2];
    for (int i = 0; i < 2; ++i)
        aP[i] = A + (size_t)(m0 + i * 16 + r) * lda + hf * 8;
    const __bf16* bP[4];
    for (int j = 0; j < 4; ++j)
        bP[j] = W + (size_t)(n0 + j * 16 + r) * ldb + hf * 8;

    for (int k0 = 0; k0 < K; k0 += 32) {
        AFrag a[2];
        load_a(a[0], aP[0] + k0);
        load_a(a[1], aP[1] + k0);
        BFrag b[4];
        #pragma unroll
        for (int j = 0; j < 4; ++j) load_b(b[j], bP[j] + k0);
        #pragma unroll
        for (int i = 0; i < 2; ++i)
            #pragma unroll
            for (int j = 0; j < 4; ++j)
                acc[i][j] = wmma_bf16(a[i], b[j], acc[i][j]);
    }

    for (int i = 0; i < 2; ++i)
        for (int j = 0; j < 4; ++j)
            #pragma unroll
            for (int v = 0; v < 8; ++v) {
                const int row = m0 + i * 16 + hf * 8 + v;
                const int col = n0 + j * 16 + r;
                float val = acc[i][j][v] + bias[col];
                if (col < DII) xssm[(size_t)row * DII + col] = (__bf16)val;
                else           gate[(size_t)row * DII + (col - DII)] = val;
            }
}

// ---------------- GEMM 2: conv1d as 4 shifted GEMMs, async-LDS staged ----------------
// A: xssm bf16 [4096 x 2048]; W: wcv bf16 [4][2048 x 2048]
__global__ __launch_bounds__(256) void k_gemm_conv(const __bf16* __restrict__ A,
                                                   const __bf16* __restrict__ W,
                                                   const float* __restrict__ bias,
                                                   float* __restrict__ xconv) {
    __shared__ __bf16 sA[2][128 * LPITCH];   // 10 KB each
    __shared__ __bf16 sB[2][128 * LPITCH];

    const int tid = threadIdx.x, wave = tid >> 5, lane = tid & 31;
    const int r = lane & 15, hf = lane >> 4;
    const int m0 = blockIdx.y * 128 + (wave >> 1) * 32;
    const int n0 = blockIdx.x * 128 + (wave & 1) * 64;
    const int mB = blockIdx.y * 128;         // block row base (tokens)
    const int nB = blockIdx.x * 128;         // block col base (out channels)
    const bool seqStart = (mB & (LL - 1)) == 0;

    v8f acc[2][4];
    for (int i = 0; i < 2; ++i) for (int j = 0; j < 4; ++j) acc[i][j] = v8f_zero();

    // stage s in [0, 256): kc = s>>6, k0 = (s&63)*32
    auto issue = [&](int s, int buf) {
        const int kc = s >> 6;
        const int k0 = (s & 63) << 5;
        const int shift = kc - (KCC - 1);
        const __bf16* wk = W + (size_t)kc * DII * DII;
        #pragma unroll
        for (int q = 0; q < 2; ++q) {
            const int c = tid * 2 + q;       // 512 16B chunks per tile
            const int row = c >> 2, cc = c & 3;
            async_cp16(&sA[buf][row * LPITCH + cc * 8],
                       A + (size_t)(mB + row + shift) * DII + k0 + cc * 8);
            async_cp16(&sB[buf][row * LPITCH + cc * 8],
                       wk + (size_t)(nB + row) * DII + k0 + cc * 8);
        }
    };

    issue(0, 0);
    const int S = KCC * (DII / 32);          // 256 stages
    for (int s = 0; s < S; ++s) {
        const int cur = s & 1;
        const bool more = (s + 1) < S;
        if (more) issue(s + 1, (s + 1) & 1);
        if (more) async_wait_le4(); else async_wait_le0();
        __syncthreads();                      // this stage's tiles visible

        // zero rows that fall before the sequence start (causal left-pad)
        const int kc = s >> 6;
        const int nz = (seqStart && kc < KCC - 1) ? (KCC - 1 - kc) : 0;
        if (nz > 0) {
            if (tid < nz * LPITCH)
                sA[cur][(tid / LPITCH) * LPITCH + (tid % LPITCH)] = (__bf16)0.0f;
            __syncthreads();
        }

        AFrag a[2];
        #pragma unroll
        for (int i = 0; i < 2; ++i) {
            const __bf16* p = &sA[cur][((wave >> 1) * 32 + i * 16 + r) * LPITCH + hf * 8];
            a[i].q[0] = *reinterpret_cast<const uint4*>(p);
            a[i].q[1] = *reinterpret_cast<const uint4*>(p + 16);
        }
        BFrag b[4];
        #pragma unroll
        for (int j = 0; j < 4; ++j) {
            const __bf16* p = &sB[cur][((wave & 1) * 64 + j * 16 + r) * LPITCH + hf * 8];
            b[j].q[0] = *reinterpret_cast<const uint4*>(p);
            b[j].q[1] = *reinterpret_cast<const uint4*>(p + 16);
        }
        #pragma unroll
        for (int i = 0; i < 2; ++i)
            #pragma unroll
            for (int j = 0; j < 4; ++j)
                acc[i][j] = wmma_bf16(a[i], b[j], acc[i][j]);

        __syncthreads();                      // done reading before buffer reuse
    }

    for (int i = 0; i < 2; ++i)
        for (int j = 0; j < 4; ++j)
            #pragma unroll
            for (int v = 0; v < 8; ++v) {
                const int row = m0 + i * 16 + hf * 8 + v;
                const int col = n0 + j * 16 + r;
                float val = acc[i][j][v] + bias[col];
                xconv[(size_t)row * DII + col] = silu_f(val);
            }
}

// ---------------- SSM scan (exact recurrence) + D skip + gate ----------------
__global__ __launch_bounds__(256) void k_scan(const float* __restrict__ xconv,
                                              const float* __restrict__ gate,
                                              const float* __restrict__ A_log,
                                              const float* __restrict__ dtp,
                                              const float* __restrict__ Bp,
                                              const float* __restrict__ Cp,
                                              const float* __restrict__ Dp,
                                              __bf16* __restrict__ h) {
    const int idx = blockIdx.x * 256 + threadIdx.x;   // 0..8191 = (b,d)
    const int b = idx >> 11, d = idx & (DII - 1);
    const float dts = 1.f / (1.f + expf(-dtp[d]));
    float a[NST], bb[NST], c[NST], s[NST];
    #pragma unroll
    for (int n = 0; n < NST; ++n) {
        const float Av = -expf(A_log[d * NST + n]);
        a[n]  = expf(dts * Av);
        bb[n] = (a[n] - 1.f) / (Av + 1e-8f) * Bp[d * NST + n];
        c[n]  = Cp[d * NST + n];
        s[n]  = 0.f;
    }
    const float Dv = Dp[d];
    const size_t base = (size_t)b * LL * DII + d;
    for (int t = 0; t < LL; ++t) {
        const size_t off = base + (size_t)t * DII;
        const float xv = xconv[off];
        float y = Dv * xv;
        #pragma unroll
        for (int n = 0; n < NST; ++n) {
            s[n] = fmaf(a[n], s[n], bb[n] * xv);
            y = fmaf(c[n], s[n], y);
        }
        const float g = gate[off];
        h[off] = (__bf16)(y * silu_f(g));
    }
}

// ---------------- GEMM 3: out = h @ out_proj_w^T + b + residual ----------------
__global__ __launch_bounds__(256) void k_gemm_out(const __bf16* __restrict__ A,
                                                  const __bf16* __restrict__ W,
                                                  const float* __restrict__ bias,
                                                  const float* __restrict__ resid,
                                                  float* __restrict__ out) {
    const int lda = DII, ldb = DII, K = DII;
    const int tid = threadIdx.x, wave = tid >> 5, lane = tid & 31;
    const int r = lane & 15, hf = lane >> 4;
    const int m0 = blockIdx.y * 128 + (wave >> 1) * 32;
    const int n0 = blockIdx.x * 128 + (wave & 1) * 64;

    v8f acc[2][4];
    for (int i = 0; i < 2; ++i) for (int j = 0; j < 4; ++j) acc[i][j] = v8f_zero();

    const __bf16* aP[2];
    for (int i = 0; i < 2; ++i)
        aP[i] = A + (size_t)(m0 + i * 16 + r) * lda + hf * 8;
    const __bf16* bP[4];
    for (int j = 0; j < 4; ++j)
        bP[j] = W + (size_t)(n0 + j * 16 + r) * ldb + hf * 8;

    for (int k0 = 0; k0 < K; k0 += 32) {
        AFrag a[2];
        load_a(a[0], aP[0] + k0);
        load_a(a[1], aP[1] + k0);
        BFrag b[4];
        #pragma unroll
        for (int j = 0; j < 4; ++j) load_b(b[j], bP[j] + k0);
        #pragma unroll
        for (int i = 0; i < 2; ++i)
            #pragma unroll
            for (int j = 0; j < 4; ++j)
                acc[i][j] = wmma_bf16(a[i], b[j], acc[i][j]);
    }

    for (int i = 0; i < 2; ++i)
        for (int j = 0; j < 4; ++j)
            #pragma unroll
            for (int v = 0; v < 8; ++v) {
                const int row = m0 + i * 16 + hf * 8 + v;
                const int col = n0 + j * 16 + r;
                const size_t o = (size_t)row * DMM + col;
                out[o] = acc[i][j][v] + bias[col] + resid[o];
            }
}

// ---------------- launch ----------------
extern "C" void kernel_launch(void* const* d_in, const int* in_sizes, int n_in,
                              void* d_out, int out_size, void* d_ws, size_t ws_size,
                              hipStream_t stream) {
    const float* x      = (const float*)d_in[0];
    const float* norm_w = (const float*)d_in[1];
    const float* ipw    = (const float*)d_in[2];
    const float* ipb    = (const float*)d_in[3];
    const float* cvw    = (const float*)d_in[4];
    const float* cvb    = (const float*)d_in[5];
    const float* opw    = (const float*)d_in[6];
    const float* opb    = (const float*)d_in[7];
    const float* A_log  = (const float*)d_in[8];
    const float* Dp     = (const float*)d_in[9];
    const float* dtp    = (const float*)d_in[10];
    const float* Bp     = (const float*)d_in[11];
    const float* Cp     = (const float*)d_in[12];
    float* out = (float*)d_out;

    char* ws = (char*)d_ws;
    size_t off = 0;
    __bf16* xn   = (__bf16*)(ws + off); off += (size_t)NTOK * DMM * 2;        // 8 MB
    __bf16* wip  = (__bf16*)(ws + off); off += (size_t)2 * DII * DMM * 2;     // 8 MB
    __bf16* wcv  = (__bf16*)(ws + off); off += (size_t)KCC * DII * DII * 2;   // 32 MB
    __bf16* wop  = (__bf16*)(ws + off); off += (size_t)DMM * DII * 2;         // 4 MB
    __bf16* xssm = (__bf16*)(ws + off); off += (size_t)NTOK * DII * 2;        // 16 MB
    float*  gate = (float*) (ws + off); off += (size_t)NTOK * DII * 4;        // 32 MB
    float*  xcv  = (float*) (ws + off); off += (size_t)NTOK * DII * 4;        // 32 MB
    __bf16* hbuf = (__bf16*)(ws + off); off += (size_t)NTOK * DII * 2;        // 16 MB

    // weight casts
    k_cast_bf16<<<(2 * DII * DMM) / 256, 256, 0, stream>>>(ipw, wip, 2 * DII * DMM);
    k_convw_rearrange<<<(DII * DII) / 256, 256, 0, stream>>>(cvw, wcv);
    k_cast_bf16<<<(DMM * DII) / 256, 256, 0, stream>>>(opw, wop, DMM * DII);

    // pipeline
    k_rmsnorm<<<NTOK, 256, 0, stream>>>(x, norm_w, xn);
    k_gemm_inproj<<<dim3((2 * DII) / 128, NTOK / 128), 256, 0, stream>>>(
        xn, wip, ipb, xssm, gate);
    k_gemm_conv<<<dim3(DII / 128, NTOK / 128), 256, 0, stream>>>(
        xssm, wcv, cvb, xcv);
    k_scan<<<(BB * DII) / 256, 256, 0, stream>>>(
        xcv, gate, A_log, dtp, Bp, Cp, Dp, hbuf);
    k_gemm_out<<<dim3(DMM / 128, NTOK / 128), 256, 0, stream>>>(
        hbuf, wop, opb, x, out);
}
